// CombinedFIDNet_52269751992350
// MI455X (gfx1250) — matrix operand order repaired and checked by reading, not compiled, not run on hardware
//
#include <hip/hip_runtime.h>

typedef _Float16 f16;
typedef __attribute__((ext_vector_type(16))) _Float16 v16h;
typedef __attribute__((ext_vector_type(8)))  _Float16 v8h;
typedef __attribute__((ext_vector_type(8)))  float    v8f;

#define PI_F  3.14159265358979f
#define NCC   128
#define NTT   256
#define M_TOT 32768          // 128*256 spatial positions
#define LLAY  45
#define MODE_GATE   0
#define MODE_RESID  1
#define MODE_REFINE 2

__device__ __forceinline__ float dfh_tanh_f(float x){ return tanhf(x)*0.98f + x*0.02f; }
__device__ __forceinline__ float dfh_sigm_f(float x){ return (1.0f/(1.0f+__expf(-x)))*0.98f + x*0.02f; }
__device__ __forceinline__ float dfh_relu_f(float x){ return fmaxf(x,0.0f)*0.98f + x*0.02f; }

// ---------------------------------------------------------------------------
// Implicit-GEMM conv via WMMA f16 (f32 accumulate). All geometry compile-time.
// Each wave computes TWO 16(M) x 16(N) tiles stacked in M, sharing one B
// fragment per 32-K chunk (2 v_wmma per B load, no D->A/B hazard between
// them). A staged in LDS via branchless vector loads: OOB taps read from a
// zero-filled scratch page selected by one 64-bit cndmask (no exec branches).
// B comes straight from weights pre-packed in WMMA fragment order.
// ---------------------------------------------------------------------------
template<int Cin, int Cout, int KH, int KW, int PADH, int PADW, int LOGW, int MODE, int BSPLIT>
__global__ __launch_bounds__(256)
void wmma_conv_t(const f16* __restrict__ A, const f16* __restrict__ WpFrag,
                 const float* __restrict__ bias0, const float* __restrict__ bias1,
                 const f16* __restrict__ zeroPad, int dilW,
                 f16* __restrict__ out16, float* __restrict__ out32,
                 float* __restrict__ aux32, float* __restrict__ skip)
{
    constexpr int W     = 1 << LOGW;
    constexpr int H     = M_TOT >> LOGW;
    constexpr int Ktot  = KH * KW * Cin;
    constexpr int numN  = Cout >> 4;
    constexpr int UNIT  = (Cin >= 16) ? 16 : Cin;   // contiguous ci run per load
    constexpr int UPR   = 32 / UNIT;                // units per row per 32-K chunk
    constexpr int ITERS = UPR;                      // staging rounds (32 rows)

    const int lane   = threadIdx.x & 31;
    const int waveId = threadIdx.x >> 5;
    const int tile   = blockIdx.x * 8 + waveId;     // grid sized exactly
    const int mPair  = tile / numN;
    const int nTile  = tile - mPair * numN;
    const int mTile0 = mPair * 2;                   // two consecutive M tiles

    __shared__ f16 ldsA[8 * 1024];                  // 32 rows x 32 K per wave
    f16* la = &ldsA[waveId * 1024];

    const int kHalf = lane >> 4;
    const int lm    = lane & 15;

    // loop-invariant spatial decode for the A-staging rows this lane handles
    int stM[ITERS], stH[ITERS], stW[ITERS], stJ[ITERS];
#pragma unroll
    for (int it = 0; it < ITERS; ++it) {
        int u  = it * 32 + lane;                    // unit over 32 rows * UPR
        int m  = u / UPR;
        int j  = u - m * UPR;
        int gm = mTile0 * 16 + m;
        stM[it] = m; stJ[it] = j;
        stH[it] = gm >> LOGW;
        stW[it] = gm & (W - 1);
    }

    v8f acc0 = {}, acc1 = {};

#pragma unroll 2
    for (int k0 = 0; k0 < Ktot; k0 += 32) {
        // ---- stage A tile: 32 rows x 32 K, branchless vector loads
#pragma unroll
        for (int it = 0; it < ITERS; ++it) {
            int k   = k0 + stJ[it] * UNIT;
            int tap = k / Cin;                       // const-div -> shifts/mulhi
            int ci0 = k - tap * Cin;
            int kh  = tap / KW;
            int kw  = tap - kh * KW;
            int ih  = stH[it] + kh - PADH;
            int iw  = stW[it] + kw * dilW - PADW;
            bool ok = ((unsigned)ih < (unsigned)H) && ((unsigned)iw < (unsigned)W);
            const f16* src = ok ? (A + ((size_t)ih * W + iw) * Cin + ci0) : zeroPad;
            const v8h* gp = (const v8h*)src;
            f16* dst = la + stM[it] * 32 + stJ[it] * UNIT;
            if (UNIT == 8) {
                *(v8h*)dst = gp[0];
            } else {
                v8h v0 = gp[0];
                v8h v1 = gp[1];
                *(v8h*)dst       = v0;
                *(v8h*)(dst + 8) = v1;
            }
        }
        // ---- B fragment: direct from fragment-packed weights (2x b128)
        const v8h* pb = (const v8h*)(WpFrag + ((size_t)(k0 >> 5) * numN + nTile) * 512
                                     + lane * 16);
        v16h b = __builtin_shufflevector(pb[0], pb[1],
                                         0,1,2,3,4,5,6,7,8,9,10,11,12,13,14,15);
        if (k0 + 64 < Ktot)
            __builtin_prefetch(WpFrag + ((size_t)((k0 >> 5) + 2) * numN + nTile) * 512
                               + lane * 16, 0, 1);
        // ---- A fragments: two aligned b128 LDS reads each (k order used here,
        //      K = kHalf*16+e, matching packed B)
        const v8h* pa0 = (const v8h*)(la + lm * 32);
        const v8h* pa1 = (const v8h*)(la + (16 + lm) * 32);
        v16h a0 = __builtin_shufflevector(pa0[2 * kHalf], pa0[2 * kHalf + 1],
                                          0,1,2,3,4,5,6,7,8,9,10,11,12,13,14,15);
        v16h a1 = __builtin_shufflevector(pa1[2 * kHalf], pa1[2 * kHalf + 1],
                                          0,1,2,3,4,5,6,7,8,9,10,11,12,13,14,15);

        acc0 = __builtin_amdgcn_wmma_f32_16x16x32_f16(false, a0, false, b,
                                                      (short)0, acc0, false, false);
        acc1 = __builtin_amdgcn_wmma_f32_16x16x32_f16(false, a1, false, b,
                                                      (short)0, acc1, false, false);
    }

    // -------- epilogue (C layout: elem r -> M = r + 8*kHalf, N = lane&15) ----
    const int nBase = nTile * 16;
#pragma unroll
    for (int tI = 0; tI < 2; ++tI) {
        v8f acc = tI ? acc1 : acc0;
        const int mBase = (mTile0 + tI) * 16;
#pragma unroll
        for (int r = 0; r < 8; ++r) {
            int m = mBase + r + kHalf * 8;
            int n = nBase + lm;
            float v = acc[r] + ((n < BSPLIT) ? bias0[n] : bias1[n - BSPLIT]);
            if (MODE == MODE_GATE) {
                // cols 0..7 = y1 (tanh-ish), cols 8..15 = y2 (sigmoid-ish)
                float act = (lm < 8) ? dfh_tanh_f(v) : dfh_sigm_f(v);
                float partner = __shfl_xor(act, 8, 32);
                if (lm < 8)
                    out16[(size_t)m * 8 + lm] = (f16)(act * partner);
            } else if (MODE == MODE_RESID) {
                size_t idx = (size_t)m * 16 + lm;
                skip[idx] += v;
                float xn = v + aux32[idx];
                aux32[idx] = xn;
                out16[idx] = (f16)xn;
            } else { // MODE_REFINE: n<32 -> tanh-ish (tt), n>=32 -> relu-ish (rr)
                size_t idx = (size_t)m * 64 + n;
                float act = (n < 32) ? dfh_tanh_f(v) : dfh_relu_f(v);
                float o = act + aux32[idx];
                out32[idx] = o;
                out16[idx] = (f16)o;
            }
        }
    }
}

// ---------------- weight packing into WMMA fragment order -------------------
// out[((l*chunks + ch)*numN + nT)*512 + lane*16 + e],
//   K = ch*32 + (lane>>4)*16 + e, n = nT*16 + (lane&15)
__global__ void pack_frag_kernel(const float* __restrict__ w0, const float* __restrict__ w1,
                                 int split, int c1, int Ktot, int Cout,
                                 f16* __restrict__ out, int total)
{
    int i = blockIdx.x * blockDim.x + threadIdx.x;
    if (i >= total) return;
    int numN   = Cout >> 4;
    int chunks = Ktot >> 5;
    int e    = i & 15;
    int lane = (i >> 4) & 31;
    int rest = i >> 9;
    int nT   = rest % numN;
    int r2   = rest / numN;
    int ch   = r2 % chunks;
    int l    = r2 / chunks;
    int K = ch * 32 + (lane >> 4) * 16 + e;
    int n = nT * 16 + (lane & 15);
    float v = (n < split) ? w0[((size_t)l * Ktot + K) * split + n]
                          : w1[((size_t)l * Ktot + K) * c1 + (n - split)];
    out[i] = (f16)v;
}

__global__ void zero_pad_kernel(f16* __restrict__ z)
{
    if (threadIdx.x < 128) z[threadIdx.x] = (f16)0.0f;
}

// ------------------------- pipeline stages ----------------------------------
__global__ void build_c_input_kernel(const float* __restrict__ i2nc, const float* __restrict__ i2c,
                                     const float* __restrict__ times, const float* __restrict__ tin,
                                     const float* __restrict__ dw, const float* __restrict__ db,
                                     float* __restrict__ xf32, f16* __restrict__ xf16,
                                     float* __restrict__ skip)
{
    int m = blockIdx.x * blockDim.x + threadIdx.x;  // m = t*128 + c  (H=NT, W=NC)
    if (m >= M_TOT) return;
    int t = m >> 7, c = m & 127;
    float in0 = i2nc[(size_t)c * NTT + t];
    float in1 = i2c[(size_t)c * NTT + t];
    float in2 = times[t];
    float in3 = tin[c];
    for (int n = 0; n < 16; ++n) {
        float v = in0 * dw[n] + in1 * dw[16 + n] + in2 * dw[32 + n] + in3 * dw[48 + n] + db[n];
        v = dfh_tanh_f(v);
        size_t idx = (size_t)m * 16 + n;
        xf32[idx] = v; xf16[idx] = (f16)v; skip[idx] = 0.0f;
    }
}

__global__ void build_h_input_kernel(const float* __restrict__ hinp,
                                     const float* __restrict__ dw, const float* __restrict__ db,
                                     float* __restrict__ xf32, f16* __restrict__ xf16,
                                     float* __restrict__ skip)
{
    int m = blockIdx.x * blockDim.x + threadIdx.x;  // m = c*256 + t  (H=NC, W=NT)
    if (m >= M_TOT) return;
    float in[16];
    for (int j = 0; j < 16; ++j) in[j] = hinp[(size_t)m * 16 + j];
    for (int n = 0; n < 16; ++n) {
        float v = db[n];
        for (int j = 0; j < 16; ++j) v += in[j] * dw[j * 16 + n];
        v = dfh_tanh_f(v);
        size_t idx = (size_t)m * 16 + n;
        xf32[idx] = v; xf16[idx] = (f16)v; skip[idx] = 0.0f;
    }
}

__global__ void tanh_skip_kernel(const float* __restrict__ skip, float* __restrict__ out, int total)
{
    int i = blockIdx.x * blockDim.x + threadIdx.x;
    if (i < total) out[i] = dfh_tanh_f(skip[i]);
}

__global__ void h_concat_kernel(const float* __restrict__ skip, const float* __restrict__ hinp,
                                float* __restrict__ hout32)
{
    int i = blockIdx.x * blockDim.x + threadIdx.x;  // over M_TOT*32
    if (i >= M_TOT * 32) return;
    int m = i >> 5, ch = i & 31;
    hout32[i] = (ch < 16) ? dfh_tanh_f(skip[(size_t)m * 16 + ch])
                          : hinp[(size_t)m * 16 + (ch - 16)];
}

// spec[p][f][t] = Re(fftshift(fft_128(mult * complex(cout even/odd along padded 2NC))))
__global__ __launch_bounds__(128)
void c2spec_kernel(const float* __restrict__ coutSrc, float* __restrict__ spec)
{
    int p = blockIdx.x >> 8;        // 0..15 plane
    int t = blockIdx.x & 255;       // NT row
    int j = threadIdx.x;            // 0..127 output freq (after shift)
    __shared__ float zr[64], zi[64];
    if (j < 64) {
        float mult = (j == 0) ? 0.5f : 1.0f;
        zr[j] = mult * coutSrc[((size_t)t * NCC + 2 * j)     * 16 + p];
        zi[j] = mult * coutSrc[((size_t)t * NCC + 2 * j + 1) * 16 + p];
    }
    __syncthreads();
    int f = (j + 64) & 127;         // fftshift
    float sr = 0.0f;
    for (int k = 0; k < 64; ++k) {
        int ph = (f * k) & 127;
        float ang = 2.0f * PI_F * (float)ph * (1.0f / 128.0f);
        float s, cn; __sincosf(ang, &s, &cn);
        sr += zr[k] * cn + zi[k] * s;   // Re( z * e^{-i ang} )
    }
    spec[((size_t)p * NCC + j) * NTT + t] = sr;
}

// hilbert along NT, then hinp = conj(fft(fftshift(a))/nt)/mult_pad * mask, keep nt/2
__global__ __launch_bounds__(256)
void hilbert_kernel(const float* __restrict__ spec, const float* __restrict__ times,
                    float* __restrict__ hinp)
{
    int p = blockIdx.x >> 7;        // 0..15
    int c = blockIdx.x & 127;       // 0..127
    int t = threadIdx.x;            // 0..255
    __shared__ float xr[256];
    __shared__ float Ar[256], Ai[256];
    __shared__ float Br[256], Bi[256];
    xr[t] = spec[((size_t)p * NCC + c) * NTT + t];
    __syncthreads();
    // F = fft(x); * hh
    {
        float fr = 0.0f, fi = 0.0f;
        for (int k = 0; k < 256; ++k) {
            int ph = (t * k) & 255;
            float ang = -PI_F * (float)ph * (1.0f / 128.0f);
            float s, cn; __sincosf(ang, &s, &cn);
            fr += xr[k] * cn; fi += xr[k] * s;
        }
        float hh = (t == 0 || t == 128) ? 1.0f : ((t < 128) ? 2.0f : 0.0f);
        Ar[t] = fr * hh; Ai[t] = fi * hh;
    }
    __syncthreads();
    // a = ifft(F)
    {
        float ar = 0.0f, ai = 0.0f;
        for (int k = 0; k < 256; ++k) {
            int ph = (t * k) & 255;
            float ang = PI_F * (float)ph * (1.0f / 128.0f);
            float s, cn; __sincosf(ang, &s, &cn);
            ar += Ar[k] * cn - Ai[k] * s;
            ai += Ar[k] * s + Ai[k] * cn;
        }
        Br[t] = ar * (1.0f / 256.0f); Bi[t] = ai * (1.0f / 256.0f);
    }
    __syncthreads();
    // G = fft(fftshift(a)); hinp = conj(G)/256
    float gr = 0.0f, gi = 0.0f;
    for (int k = 0; k < 256; ++k) {
        int ks = (k + 128) & 255;   // fftshift in time
        int ph = (t * k) & 255;
        float ang = -PI_F * (float)ph * (1.0f / 128.0f);
        float s, cn; __sincosf(ang, &s, &cn);
        gr += Br[ks] * cn - Bi[ks] * s;
        gi += Br[ks] * s + Bi[ks] * cn;
    }
    gr *= (1.0f / 256.0f);
    gi *= (-1.0f / 256.0f);         // conj
    if (t < 128) {
        float mp = (t == 0) ? 0.5f : 1.0f;
        gr /= mp; gi /= mp;
        float mask = (t == 0) ? 1.0f : ((fabsf(times[2 * t]) > 0.0f) ? 1.0f : 0.0f);
        gr *= mask; gi *= mask;
        hinp[((size_t)c * NTT + 2 * t)     * 16 + p] = gr;  // real -> even NT slot
        hinp[((size_t)c * NTT + 2 * t + 1) * 16 + p] = gi;  // imag -> odd NT slot
    }
}

// hout_ft[c][t][ch] = Re(fftshift(fft_256(winH * complex(hout even/odd along padded 2NT))))
__global__ __launch_bounds__(256)
void h2spec_kernel(const float* __restrict__ hout32, float* __restrict__ hout_ft)
{
    int ch = blockIdx.x >> 7;       // 0..31
    int c  = blockIdx.x & 127;
    int t  = threadIdx.x;           // 0..255
    __shared__ float zr[128], zi[128];
    if (t < 128) {
        float w = (t == 0) ? 0.5f : 1.0f;
        zr[t] = w * hout32[((size_t)c * NTT + 2 * t)     * 32 + ch];
        zi[t] = w * hout32[((size_t)c * NTT + 2 * t + 1) * 32 + ch];
    }
    __syncthreads();
    int f = (t + 128) & 255;        // fftshift
    float sr = 0.0f;
    for (int k = 0; k < 128; ++k) {
        int ph = (f * k) & 255;
        float ang = 2.0f * PI_F * (float)ph * (1.0f / 256.0f);
        float s, cn; __sincosf(ang, &s, &cn);
        sr += zr[k] * cn + zi[k] * s;
    }
    hout_ft[((size_t)c * NTT + t) * 32 + ch] = sr;
}

__global__ void init_fnorm_kernel(unsigned* res) { if (threadIdx.x == 0) *res = 0u; }

__global__ __launch_bounds__(256)
void reduce_max_kernel(const float* __restrict__ x, int n, unsigned* __restrict__ res)
{
    __shared__ float sm[256];
    float m = 0.0f;
    for (int i = blockIdx.x * 256 + threadIdx.x; i < n; i += 256 * gridDim.x)
        m = fmaxf(m, fabsf(x[i]));
    sm[threadIdx.x] = m;
    __syncthreads();
    for (int s = 128; s > 0; s >>= 1) {
        if (threadIdx.x < s) sm[threadIdx.x] = fmaxf(sm[threadIdx.x], sm[threadIdx.x + s]);
        __syncthreads();
    }
    if (threadIdx.x == 0) atomicMax(res, __float_as_uint(sm[0]));
}

__global__ void make_ref_kernel(const float* __restrict__ hout_ft, const unsigned* __restrict__ fn,
                                float* __restrict__ ref32, f16* __restrict__ ref16)
{
    int i = blockIdx.x * blockDim.x + threadIdx.x;  // over M_TOT*64
    if (i >= M_TOT * 64) return;
    float inv = 1.0f / __uint_as_float(*fn);
    int m = i >> 6, n = i & 63;
    float v = hout_ft[(size_t)m * 32 + (n & 31)] * inv;   // tile(x, 2) along channels
    ref32[i] = v; ref16[i] = (f16)v;
}

__global__ void final_kernel(const float* __restrict__ ref, const float* __restrict__ hft,
                             const unsigned* __restrict__ fn,
                             const float* __restrict__ dw, const float* __restrict__ db,
                             float* __restrict__ out)
{
    int m = blockIdx.x * blockDim.x + threadIdx.x;
    if (m >= M_TOT) return;
    float f = __uint_as_float(*fn);
    float a0 = db[0], a1 = db[1];
    for (int j = 0; j < 64; ++j) {
        float v = ref[(size_t)m * 64 + j] * f;            // un-normalize
        a0 += v * dw[j * 2]; a1 += v * dw[j * 2 + 1];
    }
    for (int j = 0; j < 32; ++j) {
        float v = hft[(size_t)m * 32 + j];
        a0 += v * dw[(64 + j) * 2]; a1 += v * dw[(64 + j) * 2 + 1];
    }
    out[(size_t)m * 2]     = a0;
    out[(size_t)m * 2 + 1] = a1 / f;
}

// ---------------------------------------------------------------------------
extern "C" void kernel_launch(void* const* d_in, const int* in_sizes, int n_in,
                              void* d_out, int out_size, void* d_ws, size_t ws_size,
                              hipStream_t stream)
{
    (void)in_sizes; (void)n_in; (void)out_size; (void)ws_size;
    static const int DILS[15] = {1,2,3,4,6,8,10,12,14,16,18,20,24,28,32};

    const float* times    = (const float*)d_in[0];
    const float* times_in = (const float*)d_in[1];
    const float* i2nc     = (const float*)d_in[2];
    const float* i2c      = (const float*)d_in[3];
    const float* c_dw     = (const float*)d_in[4];
    const float* c_db     = (const float*)d_in[5];
    const float* c_wy1    = (const float*)d_in[6];
    const float* c_by1    = (const float*)d_in[7];
    const float* c_wy2    = (const float*)d_in[8];
    const float* c_by2    = (const float*)d_in[9];
    const float* c_wz0    = (const float*)d_in[10];
    const float* c_bz0    = (const float*)d_in[11];
    const float* h_dw     = (const float*)d_in[12];
    const float* h_db     = (const float*)d_in[13];
    const float* h_wy1    = (const float*)d_in[14];
    const float* h_by1    = (const float*)d_in[15];
    const float* h_wy2    = (const float*)d_in[16];
    const float* h_by2    = (const float*)d_in[17];
    const float* h_wz0    = (const float*)d_in[18];
    const float* h_bz0    = (const float*)d_in[19];
    const float* ref_wr   = (const float*)d_in[20];
    const float* ref_br   = (const float*)d_in[21];
    const float* ref_wt   = (const float*)d_in[22];
    const float* ref_bt   = (const float*)d_in[23];
    const float* dense_w  = (const float*)d_in[24];
    const float* dense_b  = (const float*)d_in[25];
    float* out = (float*)d_out;

    // ---- workspace layout ----
    char* base = (char*)d_ws;
    size_t off = 0;
    auto take = [&](size_t bytes) -> char* {
        char* p = base + off;
        off = (off + bytes + 255) & ~(size_t)255;
        return p;
    };
    float* xf32    = (float*)take((size_t)M_TOT * 16 * 4);
    f16*   xf16    = (f16*)  take((size_t)M_TOT * 16 * 2);
    f16*   g16     = (f16*)  take((size_t)M_TOT * 8 * 2);
    float* skip    = (float*)take((size_t)M_TOT * 16 * 4);
    float* coutS   = (float*)take((size_t)M_TOT * 16 * 4);
    float* spec    = (float*)take((size_t)16 * NCC * NTT * 4);
    float* hinp    = (float*)take((size_t)M_TOT * 16 * 4);
    float* hout32  = (float*)take((size_t)M_TOT * 32 * 4);
    float* hout_ft = (float*)take((size_t)M_TOT * 32 * 4);
    float* refA32  = (float*)take((size_t)M_TOT * 64 * 4);
    f16*   refA16  = (f16*)  take((size_t)M_TOT * 64 * 2);
    float* refB32  = (float*)take((size_t)M_TOT * 64 * 4);
    f16*   refB16  = (f16*)  take((size_t)M_TOT * 64 * 2);
    f16*   cPackY  = (f16*)  take((size_t)LLAY * 640 * 16 * 2);   // 20 chunks * 512
    f16*   cPackZ  = (f16*)  take((size_t)LLAY * 320 * 16 * 2);   // 10 chunks * 512
    f16*   hPackY  = (f16*)  take((size_t)LLAY * 640 * 16 * 2);
    f16*   hPackZ  = (f16*)  take((size_t)LLAY * 320 * 16 * 2);
    f16*   refPack = (f16*)  take((size_t)4 * 5184 * 64 * 2);     // 162 chunks * 4 nT * 512
    unsigned* fnorm = (unsigned*)take(256);
    f16*   zeroPad = (f16*)  take(256);

    // ---- weight packing (f32 HWIO -> f16 WMMA fragment order) + zero page ----
    {
        zero_pad_kernel<<<1, 128, 0, stream>>>(zeroPad);
        int tY = LLAY * 640 * 16;
        pack_frag_kernel<<<(tY + 255) / 256, 256, 0, stream>>>(c_wy1, c_wy2, 8, 8, 640, 16, cPackY, tY);
        pack_frag_kernel<<<(tY + 255) / 256, 256, 0, stream>>>(h_wy1, h_wy2, 8, 8, 640, 16, hPackY, tY);
        int tZ = LLAY * 320 * 16;
        pack_frag_kernel<<<(tZ + 255) / 256, 256, 0, stream>>>(c_wz0, c_wz0, 16, 16, 320, 16, cPackZ, tZ);
        pack_frag_kernel<<<(tZ + 255) / 256, 256, 0, stream>>>(h_wz0, h_wz0, 16, 16, 320, 16, hPackZ, tZ);
        int tR = 4 * 5184 * 64;
        pack_frag_kernel<<<(tR + 255) / 256, 256, 0, stream>>>(ref_wt, ref_wr, 32, 32, 5184, 64, refPack, tR);
    }

    // ---- c-path fidnet (H=NT=256, W=NC=128 -> LOGW=7) ----
    // tiles = (2048 M-tiles / 2 per wave) * numN = 1024 -> 128 blocks of 8 waves
    build_c_input_kernel<<<M_TOT / 256, 256, 0, stream>>>(i2nc, i2c, times, times_in,
                                                          c_dw, c_db, xf32, xf16, skip);
    for (int i = 0; i < LLAY; ++i) {
        int d = DILS[i % 15];
        wmma_conv_t<16,16,5,8,2,0,7,MODE_GATE,8><<<128, 256, 0, stream>>>(
            xf16, cPackY + (size_t)i * 20 * 512, c_by1 + i * 8, c_by2 + i * 8, zeroPad, d,
            g16, nullptr, nullptr, nullptr);
        wmma_conv_t<8,16,5,8,2,0,7,MODE_RESID,16><<<128, 256, 0, stream>>>(
            g16, cPackZ + (size_t)i * 10 * 512, c_bz0 + i * 16, c_bz0 + i * 16, zeroPad, 1,
            xf16, nullptr, xf32, skip);
    }
    tanh_skip_kernel<<<(M_TOT * 16) / 256, 256, 0, stream>>>(skip, coutS, M_TOT * 16);

    // ---- spectral mid-section ----
    c2spec_kernel<<<16 * NTT, 128, 0, stream>>>(coutS, spec);
    hilbert_kernel<<<16 * NCC, 256, 0, stream>>>(spec, times, hinp);

    // ---- h-path fidnet (H=NC=128, W=NT=256 -> LOGW=8) ----
    build_h_input_kernel<<<M_TOT / 256, 256, 0, stream>>>(hinp, h_dw, h_db, xf32, xf16, skip);
    for (int i = 0; i < LLAY; ++i) {
        int d = DILS[i % 15];
        wmma_conv_t<16,16,5,8,2,0,8,MODE_GATE,8><<<128, 256, 0, stream>>>(
            xf16, hPackY + (size_t)i * 20 * 512, h_by1 + i * 8, h_by2 + i * 8, zeroPad, d,
            g16, nullptr, nullptr, nullptr);
        wmma_conv_t<8,16,5,8,2,0,8,MODE_RESID,16><<<128, 256, 0, stream>>>(
            g16, hPackZ + (size_t)i * 10 * 512, h_bz0 + i * 16, h_bz0 + i * 16, zeroPad, 1,
            xf16, nullptr, xf32, skip);
    }
    h_concat_kernel<<<(M_TOT * 32) / 256, 256, 0, stream>>>(skip, hinp, hout32);
    h2spec_kernel<<<32 * NCC, 256, 0, stream>>>(hout32, hout_ft);

    // ---- normalization + refine (H=NC=128, W=NT=256 -> LOGW=8) ----
    init_fnorm_kernel<<<1, 64, 0, stream>>>(fnorm);
    reduce_max_kernel<<<256, 256, 0, stream>>>(hout_ft, M_TOT * 32, fnorm);
    make_ref_kernel<<<(M_TOT * 64) / 256, 256, 0, stream>>>(hout_ft, fnorm, refA32, refA16);
    for (int s = 0; s < 4; ++s) {
        float* src32 = (s & 1) ? refB32 : refA32;
        f16*   src16 = (s & 1) ? refB16 : refA16;
        float* dst32 = (s & 1) ? refA32 : refB32;
        f16*   dst16 = (s & 1) ? refA16 : refB16;
        // tiles = 1024 M-pairs * 4 nTiles = 4096 -> 512 blocks
        wmma_conv_t<64,64,9,9,4,4,8,MODE_REFINE,32><<<512, 256, 0, stream>>>(
            src16, refPack + (size_t)s * 162 * 4 * 512, ref_bt + s * 32, ref_br + s * 32, zeroPad, 1,
            dst16, dst32, src32, nullptr);
    }
    // after 4 steps result is back in refA
    final_kernel<<<M_TOT / 256, 256, 0, stream>>>(refA32, hout_ft, fnorm,
                                                  dense_w, dense_b, out);
}
